// Seq2SeqDecoder_28535762715221
// MI455X (gfx1250) — matrix operand order repaired and checked by reading, not compiled, not run on hardware
//
#include <hip/hip_runtime.h>

// ---------------------------------------------------------------------------
// Problem constants (from reference): B=64, SE=128, T=64, S=1024, D=1024, V=32000
// ---------------------------------------------------------------------------
#define CB  64
#define CSE 128
#define CT  64
#define CS  1024
#define CD  1024
#define CV  32000

typedef __attribute__((ext_vector_type(16))) __bf16 v16bf;
typedef __attribute__((ext_vector_type(8)))  __bf16 v8bf;
typedef __attribute__((ext_vector_type(8)))  float  v8f;

__device__ __forceinline__ __bf16 f2bf(float f) {
  unsigned u = __builtin_bit_cast(unsigned, f);
  unsigned r = (u + 0x7FFFu + ((u >> 16) & 1u)) >> 16;   // round-to-nearest-even
  unsigned short s = (unsigned short)r;
  return __builtin_bit_cast(__bf16, s);
}
__device__ __forceinline__ float sigm(float x) { return 1.0f / (1.0f + __expf(-x)); }

// ---------------------------------------------------------------------------
// WMMA GEMM:  C[M,N] = A[M,K](bf16,row-major) * B[K,N]  with B stored as
// Bt[N,K] (bf16, N-major so each lane's K-fragment is contiguous).
// One wave computes a (16*MT) x (16*NT) tile block.
//   <1,4>: 16x64 strip  (many independent waves; small B matrices)
//   <4,1>: 64x16 strip  (covers all M rows -> B streamed exactly once; used
//                        for the classifier GEMM whose B stream dominates L2)
// The K-loop is explicitly software-pipelined (double-buffered fragments) so
// loads for iteration kk+32 are outstanding while the wmmas of kk execute.
// M multiple of 16*MT, N multiple of 16*NT, K multiple of 64.
// ---------------------------------------------------------------------------
template <int MT, int NT>
__global__ __launch_bounds__(32)
void wmma_gemm_bf16(const __bf16* __restrict__ A,
                    const __bf16* __restrict__ Bt,
                    const float*  __restrict__ bias,   // [N] or nullptr
                    float*        __restrict__ Cf,     // [M,N] or nullptr
                    __bf16*       __restrict__ Cbf,    // [M,N] or nullptr
                    int M, int N, int K)
{
  const int lane = threadIdx.x;              // 0..31
  const int tn   = blockIdx.x * (16 * NT);
  const int tm   = blockIdx.y * (16 * MT);
  const int h    = lane >> 4;                // half-wave
  const int l15  = lane & 15;

  const __bf16* arow[MT];
#pragma unroll
  for (int mi = 0; mi < MT; ++mi)
    arow[mi] = A + (size_t)(tm + mi * 16 + l15) * K + h * 16;
  const __bf16* brow[NT];
#pragma unroll
  for (int ni = 0; ni < NT; ++ni)
    brow[ni] = Bt + (size_t)(tn + ni * 16 + l15) * K + h * 16;

  v8f acc[MT][NT];
#pragma unroll
  for (int mi = 0; mi < MT; ++mi)
#pragma unroll
    for (int ni = 0; ni < NT; ++ni)
      acc[mi][ni] = (v8f){};

  // ---- prologue: load fragments for kk = 0 ------------------------------
  // A fragment 16x32 bf16: lane = row; elems 0..7 -> K = kk + h*16 + i,
  // elems 8..15 -> K = kk + h*16 + 8 + i  (same contiguous order as B)
  // B fragment 32x16 bf16: lane = col; elem i -> K = kk + h*16 + i
  v8bf ca0[MT], ca1[MT];
  v16bf cb[NT];
#pragma unroll
  for (int mi = 0; mi < MT; ++mi) {
    ca0[mi] = *(const v8bf*)(arow[mi]);
    ca1[mi] = *(const v8bf*)(arow[mi] + 8);
  }
#pragma unroll
  for (int ni = 0; ni < NT; ++ni)
    cb[ni] = *(const v16bf*)(brow[ni]);

  // ---- steady state: prefetch kk+32 while computing kk -------------------
#pragma unroll 2
  for (int kk = 0; kk < K - 32; kk += 32) {
    v8bf na0[MT], na1[MT];
    v16bf nb[NT];
#pragma unroll
    for (int mi = 0; mi < MT; ++mi) {
      na0[mi] = *(const v8bf*)(arow[mi] + kk + 32);
      na1[mi] = *(const v8bf*)(arow[mi] + kk + 40);
    }
#pragma unroll
    for (int ni = 0; ni < NT; ++ni)
      nb[ni] = *(const v16bf*)(brow[ni] + kk + 32);

    v16bf a[MT];
#pragma unroll
    for (int mi = 0; mi < MT; ++mi)
#pragma unroll
      for (int i = 0; i < 8; ++i) { a[mi][i] = ca0[mi][i]; a[mi][i + 8] = ca1[mi][i]; }

#pragma unroll
    for (int mi = 0; mi < MT; ++mi)
#pragma unroll
      for (int ni = 0; ni < NT; ++ni)
        acc[mi][ni] = __builtin_amdgcn_wmma_f32_16x16x32_bf16(
            false, a[mi], false, cb[ni], (short)0, acc[mi][ni], false, false);

#pragma unroll
    for (int mi = 0; mi < MT; ++mi) { ca0[mi] = na0[mi]; ca1[mi] = na1[mi]; }
#pragma unroll
    for (int ni = 0; ni < NT; ++ni) cb[ni] = nb[ni];
  }

  // ---- epilogue iteration (kk = K-32) ------------------------------------
  {
    v16bf a[MT];
#pragma unroll
    for (int mi = 0; mi < MT; ++mi)
#pragma unroll
      for (int i = 0; i < 8; ++i) { a[mi][i] = ca0[mi][i]; a[mi][i + 8] = ca1[mi][i]; }
#pragma unroll
    for (int mi = 0; mi < MT; ++mi)
#pragma unroll
      for (int ni = 0; ni < NT; ++ni)
        acc[mi][ni] = __builtin_amdgcn_wmma_f32_16x16x32_bf16(
            false, a[mi], false, cb[ni], (short)0, acc[mi][ni], false, false);
  }

  // C/D layout: VGPR r -> row base + r + 8*h, col = base + l15
#pragma unroll
  for (int mi = 0; mi < MT; ++mi) {
#pragma unroll
    for (int ni = 0; ni < NT; ++ni) {
      const int col = tn + ni * 16 + l15;
      const float bv = bias ? bias[col] : 0.0f;
#pragma unroll
      for (int r = 0; r < 8; ++r) {
        const int row = tm + mi * 16 + r + 8 * h;
        const float v = acc[mi][ni][r] + bv;
        if (Cf)  Cf[(size_t)row * N + col] = v;
        if (Cbf) Cbf[(size_t)row * N + col] = f2bf(v);
      }
    }
  }
}

// ---------------------------------------------------------------------------
// Prep kernels
// ---------------------------------------------------------------------------

// v[k] = sum_j ayi_w[k,j] * aw_w[j]   (attention collapses to a matvec)
__global__ void compute_v(const float* __restrict__ ayi_w,
                          const float* __restrict__ aw_w,
                          float* __restrict__ v)
{
  int k = blockIdx.x * blockDim.x + threadIdx.x;
  if (k >= CS) return;
  float s = 0.0f;
  const float* row = ayi_w + (size_t)k * CS;
  for (int j = 0; j < CS; ++j) s += row[j] * aw_w[j];
  v[k] = s;
}

// attn[b,:] = softmax_se( encoder_yts[b,se,:] . v )  -- T-invariant.
// Also replicate into d_out attns for every t.
__global__ void attn_kernel(const float* __restrict__ yts,
                            const float* __restrict__ v,
                            float* __restrict__ attn_ws,
                            float* __restrict__ d_attn)
{
  const int b = blockIdx.x, se = threadIdx.x;          // 128 threads
  const float* row = yts + ((size_t)b * CSE + se) * CS;
  float s = 0.0f;
  for (int k = 0; k < CS; ++k) s += row[k] * v[k];
  __shared__ float red[CSE];
  red[se] = s; __syncthreads();
  for (int st = 64; st > 0; st >>= 1) {
    if (se < st) red[se] = fmaxf(red[se], red[se + st]);
    __syncthreads();
  }
  const float m = red[0]; __syncthreads();
  const float ex = __expf(s - m);
  red[se] = ex; __syncthreads();
  for (int st = 64; st > 0; st >>= 1) {
    if (se < st) red[se] += red[se + st];
    __syncthreads();
  }
  const float a = ex / red[0];
  attn_ws[b * CSE + se] = a;
  for (int t = 0; t < CT; ++t)
    d_attn[((size_t)b * CT + t) * CSE + se] = a;
}

// ctx[b,d] = sum_se attn[b,se] * encoder_yts[b,se,d]
__global__ void ctx_kernel(const float* __restrict__ yts,
                           const float* __restrict__ attn,
                           float* __restrict__ ctx)
{
  const int b = blockIdx.y;
  const int d = blockIdx.x * blockDim.x + threadIdx.x;
  if (d >= CS) return;
  const float* base = yts + (size_t)b * CSE * CS + d;
  const float* arow = attn + b * CSE;
  float s = 0.0f;
  for (int se = 0; se < CSE; ++se) s += arow[se] * base[(size_t)se * CS];
  ctx[b * CS + d] = s;
}

// Concatenated gate weight, transposed + bf16: wcatT[n, k], n in [0,4S), k in [0,2S)
// col n = gate g*S + j ; rows 0..S-1 from W_g, rows S..2S-1 from U_g.
__global__ void prep_wcat(const float* __restrict__ wf, const float* __restrict__ uf,
                          const float* __restrict__ wi, const float* __restrict__ ui,
                          const float* __restrict__ wo, const float* __restrict__ uo,
                          const float* __restrict__ wc, const float* __restrict__ uc,
                          __bf16* __restrict__ wcatT)
{
  size_t idx = (size_t)blockIdx.x * blockDim.x + threadIdx.x;
  const size_t total = (size_t)4 * CS * 2 * CS;
  if (idx >= total) return;
  const int K2 = 2 * CS;
  const int n = (int)(idx / K2);
  const int k = (int)(idx % K2);
  const int g = n / CS, j = n % CS;
  const float* Wg[4] = {wf, wi, wo, wc};
  const float* Ug[4] = {uf, ui, uo, uc};
  const float val = (k < CS) ? Wg[g][(size_t)k * CS + j]
                             : Ug[g][(size_t)(k - CS) * CS + j];
  wcatT[idx] = f2bf(val);
}

// Generic transpose + bf16 convert: W[K,N] -> Wt[N,K]
__global__ void prep_wT(const float* __restrict__ W, __bf16* __restrict__ Wt,
                        int K, int N)
{
  size_t idx = (size_t)blockIdx.x * blockDim.x + threadIdx.x;
  if (idx >= (size_t)K * N) return;
  const int n = (int)(idx / K);
  const int k = (int)(idx % K);
  Wt[idx] = f2bf(W[(size_t)k * N + n]);
}

__global__ void prep_bcat(const float* __restrict__ wfb, const float* __restrict__ ufb,
                          const float* __restrict__ wib, const float* __restrict__ uib,
                          const float* __restrict__ wob, const float* __restrict__ uob,
                          const float* __restrict__ wcb, const float* __restrict__ ucb,
                          float* __restrict__ bcat)
{
  int idx = blockIdx.x * blockDim.x + threadIdx.x;
  if (idx >= 4 * CS) return;
  const int g = idx / CS, j = idx % CS;
  const float* Wb[4] = {wfb, wib, wob, wcb};
  const float* Ub[4] = {ufb, uib, uob, ucb};
  bcat[idx] = Wb[g][j] + Ub[g][j];
}

__global__ void init_ht(const float* __restrict__ h, __bf16* __restrict__ ht, int n)
{
  int i = blockIdx.x * blockDim.x + threadIdx.x;
  if (i < n) ht[i] = f2bf(h[i]);
}

// ---------------------------------------------------------------------------
// Per-step kernels
// ---------------------------------------------------------------------------

// A_step[m, 0:S] = ht_bf16[m, :] ; A_step[m, S:2S] = bf16(emb[tgt[m,t], :])
__global__ void build_A(const __bf16* __restrict__ ht, const int* __restrict__ tgt,
                        const float* __restrict__ emb, __bf16* __restrict__ Astep, int t)
{
  int idx = blockIdx.x * blockDim.x + threadIdx.x;
  const int K2 = 2 * CS;
  if (idx >= CB * K2) return;
  const int m = idx / K2, k = idx % K2;
  if (k < CS) {
    Astep[idx] = ht[m * CS + k];
  } else {
    const int tok = tgt[m * CT + t];
    Astep[idx] = f2bf(emb[(size_t)tok * CD + (k - CS)]);
  }
}

// Gate combine: pre[m, g*S+j] for gates f,i,o,cbar -> ht_bf16 (next state)
__global__ void gate_combine(const float* __restrict__ pre,
                             const float* __restrict__ ctx,
                             __bf16* __restrict__ ht)
{
  int idx = blockIdx.x * blockDim.x + threadIdx.x;
  if (idx >= CB * CS) return;
  const int m = idx / CS, j = idx % CS;
  const float* p = pre + (size_t)m * 4 * CS;
  const float f  = sigm(p[j]);
  const float i  = sigm(p[CS + j]);
  const float o  = sigm(p[2 * CS + j]);
  const float cb = tanhf(p[3 * CS + j]);
  const float ct = f * ctx[idx] + i * cb;
  ht[idx] = f2bf(o * tanhf(ct));
}

// Row softmax over V=32000; writes pts[b, t, :]
__global__ void softmax_rows(const float* __restrict__ logits,
                             float* __restrict__ pts, int t)
{
  const int b = blockIdx.x;
  const float* row = logits + (size_t)b * CV;
  float* orow = pts + ((size_t)b * CT + t) * CV;
  __shared__ float red[256];
  float m = -3.4e38f;
  for (int v = threadIdx.x; v < CV; v += 256) m = fmaxf(m, row[v]);
  red[threadIdx.x] = m; __syncthreads();
  for (int s = 128; s > 0; s >>= 1) {
    if (threadIdx.x < s) red[threadIdx.x] = fmaxf(red[threadIdx.x], red[threadIdx.x + s]);
    __syncthreads();
  }
  m = red[0]; __syncthreads();
  float sum = 0.0f;
  for (int v = threadIdx.x; v < CV; v += 256) sum += __expf(row[v] - m);
  red[threadIdx.x] = sum; __syncthreads();
  for (int s = 128; s > 0; s >>= 1) {
    if (threadIdx.x < s) red[threadIdx.x] += red[threadIdx.x + s];
    __syncthreads();
  }
  const float inv = 1.0f / red[0];
  for (int v = threadIdx.x; v < CV; v += 256) orow[v] = __expf(row[v] - m) * inv;
}

// ---------------------------------------------------------------------------
// Launch
// ---------------------------------------------------------------------------
extern "C" void kernel_launch(void* const* d_in, const int* in_sizes, int n_in,
                              void* d_out, int out_size, void* d_ws, size_t ws_size,
                              hipStream_t stream)
{
  (void)in_sizes; (void)n_in; (void)out_size; (void)ws_size;

  const float* yts   = (const float*)d_in[0];
  const float* enc_h = (const float*)d_in[1];
  const int*   tgt   = (const int*)  d_in[2];
  const float* emb   = (const float*)d_in[3];
  // d_in[4]=asi_w, [5]=asi_b  -> cancel in softmax (shift invariance)
  const float* ayi_w = (const float*)d_in[6];
  // d_in[7]=ayi_b, [9]=aw_b   -> cancel in softmax
  const float* aw_w  = (const float*)d_in[8];
  const float* wf_w = (const float*)d_in[10], *wf_b = (const float*)d_in[11];
  const float* uf_w = (const float*)d_in[12], *uf_b = (const float*)d_in[13];
  const float* wi_w = (const float*)d_in[14], *wi_b = (const float*)d_in[15];
  const float* ui_w = (const float*)d_in[16], *ui_b = (const float*)d_in[17];
  const float* wo_w = (const float*)d_in[18], *wo_b = (const float*)d_in[19];
  const float* uo_w = (const float*)d_in[20], *uo_b = (const float*)d_in[21];
  const float* wc_w = (const float*)d_in[22], *wc_b = (const float*)d_in[23];
  const float* uc_w = (const float*)d_in[24], *uc_b = (const float*)d_in[25];
  const float* xh_w = (const float*)d_in[26], *xh_b = (const float*)d_in[27];
  const float* cls_w = (const float*)d_in[28], *cls_b = (const float*)d_in[29];

  float* d_pts  = (float*)d_out;                               // [B,T,V]
  float* d_attn = d_pts + (size_t)CB * CT * CV;                // [B,T,SE]

  // ---- workspace layout -------------------------------------------------
  char* ws = (char*)d_ws;
  size_t off = 0;
  auto alloc = [&](size_t bytes) -> char* {
    char* p = ws + off;
    off = (off + bytes + 255) & ~(size_t)255;
    return p;
  };
  __bf16* wcatT   = (__bf16*)alloc((size_t)4 * CS * 2 * CS * sizeof(__bf16)); // 16 MB
  __bf16* xhT     = (__bf16*)alloc((size_t)CD * CS * sizeof(__bf16));         // 2 MB
  __bf16* clsT    = (__bf16*)alloc((size_t)CV * CD * sizeof(__bf16));         // 64 MB
  float*  v_ws    = (float*) alloc(CS * sizeof(float));
  float*  attn_ws = (float*) alloc(CB * CSE * sizeof(float));
  float*  ctx_ws  = (float*) alloc((size_t)CB * CS * sizeof(float));
  float*  bcat    = (float*) alloc(4 * CS * sizeof(float));
  __bf16* A_ws    = (__bf16*)alloc((size_t)CB * 2 * CS * sizeof(__bf16));
  float*  pre_ws  = (float*) alloc((size_t)CB * 4 * CS * sizeof(float));
  __bf16* ht_bf   = (__bf16*)alloc((size_t)CB * CS * sizeof(__bf16));
  __bf16* hx_bf   = (__bf16*)alloc((size_t)CB * CD * sizeof(__bf16));
  float*  logits  = (float*) alloc((size_t)CB * CV * sizeof(float));          // 8 MB

  // ---- one-time prep ----------------------------------------------------
  compute_v<<<(CS + 255) / 256, 256, 0, stream>>>(ayi_w, aw_w, v_ws);
  attn_kernel<<<CB, CSE, 0, stream>>>(yts, v_ws, attn_ws, d_attn);
  ctx_kernel<<<dim3(CS / 256, CB), 256, 0, stream>>>(yts, attn_ws, ctx_ws);

  {
    size_t total = (size_t)4 * CS * 2 * CS;
    prep_wcat<<<(unsigned)((total + 255) / 256), 256, 0, stream>>>(
        wf_w, uf_w, wi_w, ui_w, wo_w, uo_w, wc_w, uc_w, wcatT);
  }
  prep_wT<<<(unsigned)(((size_t)CS * CD + 255) / 256), 256, 0, stream>>>(xh_w, xhT, CS, CD);
  prep_wT<<<(unsigned)(((size_t)CD * CV + 255) / 256), 256, 0, stream>>>(cls_w, clsT, CD, CV);
  prep_bcat<<<(4 * CS + 255) / 256, 256, 0, stream>>>(
      wf_b, uf_b, wi_b, ui_b, wo_b, uo_b, wc_b, uc_b, bcat);
  init_ht<<<(CB * CS + 255) / 256, 256, 0, stream>>>(enc_h, ht_bf, CB * CS);

  // ---- teacher-forced recurrence ---------------------------------------
  for (int t = 0; t < CT; ++t) {
    build_A<<<(CB * 2 * CS + 255) / 256, 256, 0, stream>>>(ht_bf, tgt, emb, A_ws, t);

    // gates preact: [64, 4096] = A[64,2048] x Wcat[2048,4096] + bcat
    // 16x64 per wave -> 256 independent waves; B strip small (16 MB)
    wmma_gemm_bf16<1, 4><<<dim3(4 * CS / 64, CB / 16), 32, 0, stream>>>(
        A_ws, wcatT, bcat, pre_ws, nullptr, CB, 4 * CS, 2 * CS);

    gate_combine<<<(CB * CS + 255) / 256, 256, 0, stream>>>(pre_ws, ctx_ws, ht_bf);

    // hx = ht @ xh_w + xh_b  ->  bf16 [64,1024]
    wmma_gemm_bf16<1, 4><<<dim3(CD / 64, CB / 16), 32, 0, stream>>>(
        ht_bf, xhT, xh_b, nullptr, hx_bf, CB, CD, CS);

    // logits = hx @ cls_w + cls_b  ->  f32 [64,32000]
    // 64x16 per wave: each wave covers ALL M rows, so clsT (64 MB, L2-resident)
    // is streamed exactly once per step instead of 4x. 2000 waves.
    wmma_gemm_bf16<4, 1><<<dim3(CV / 16, 1), 32, 0, stream>>>(
        hx_bf, clsT, cls_b, logits, nullptr, CB, CV, CD);

    softmax_rows<<<CB, 256, 0, stream>>>(logits, d_pts, t);
  }
}
